// LearnedEdge_91190745628825
// MI455X (gfx1250) — matrix operand order repaired and checked by reading, not compiled, not run on hardware
//
#include <hip/hip_runtime.h>
#include <hip/hip_bf16.h>

typedef __attribute__((ext_vector_type(16))) _Float16 v16h;
typedef __attribute__((ext_vector_type(8)))  _Float16 v8h;
typedef __attribute__((ext_vector_type(4)))  _Float16 v4h;
typedef __attribute__((ext_vector_type(8)))  float    v8f;

#define BATCH 8
#define T0C   1024
#define TAUC  32
#define FC    128
#define NODE_T (T0C + TAUC + 32)   // 1088 rows per batch in `nodes`
#define HS    136                  // padded f16 row stride (272B = 68 dwords -> +4 banks/row)

// ---------------------------------------------------------------------------
// A-tile loader: f16 row-major LDS, stride HS.
// CDNA5 16-bit A 16x32 layout: lane (m = lane&15, half = lane>>4) holds row m,
// K = {k0 + half*8 .. +7} and {k0 + 16 + half*8 .. +7}  -> two 16B loads.
// ---------------------------------------------------------------------------
__device__ inline v16h load_a_tile(const _Float16* p, int row0, int k0) {
  const int lane = threadIdx.x & 31;
  const int m = lane & 15, half = lane >> 4;
  const _Float16* base = p + (row0 + m) * HS + k0 + half * 8;
  v8h lo = *(const v8h*)(base);
  v8h hi = *(const v8h*)(base + 16);
  v16h a;
#pragma unroll
  for (int i = 0; i < 8; ++i) { a[i] = lo[i]; a[8 + i] = hi[i]; }
  return a;
}

// ---------------------------------------------------------------------------
// B-tile loader from TRANSPOSED weight staging WT[n][k] (stride HS).
// CDNA5 16-bit B 32x16 layout: lane (n = lane&15, kh = lane>>4) holds column n,
// K = {k0 + kh*8 .. +7} and {k0 + 16 + kh*8 .. +7} -> two 16B loads.
// ---------------------------------------------------------------------------
__device__ inline v16h load_b_tileT(const _Float16* p, int n0, int k0) {
  const int lane = threadIdx.x & 31;
  const int n = lane & 15, kh = lane >> 4;
  const _Float16* base = p + (n0 + n) * HS + k0 + kh * 8;
  v8h lo = *(const v8h*)(base);
  v8h hi = *(const v8h*)(base + 16);
  v16h b;
#pragma unroll
  for (int i = 0; i < 8; ++i) { b[i] = lo[i]; b[8 + i] = hi[i]; }
  return b;
}

// ===========================================================================
// Prep: transpose + convert the three 128x128 weight blocks to f16 [n][k].
//   block 0: W1a (rows 0..127 of W1, sink side)
//   block 1: W1b (rows 128..255 of W1, source side)
//   block 2: W2
// ===========================================================================
__global__ void prep_weights(const float* __restrict__ W1,
                             const float* __restrict__ W2,
                             _Float16* __restrict__ wT) {
  const float* src = (blockIdx.x == 0) ? W1
                   : (blockIdx.x == 1) ? (W1 + 128 * FC)
                   : W2;
  _Float16* dst = wT + (size_t)blockIdx.x * 128 * 128;
  for (int i = threadIdx.x; i < 128 * 128; i += 256) {
    int k = i >> 7, n = i & 127;              // coalesced read of src[i]
    dst[n * 128 + k] = (_Float16)src[i];      // scattered 2B write (one-shot, tiny)
  }
}

// ===========================================================================
// Kernel A: part = rows(nodes) @ W(128x128) [+ bias], weights pre-f16 [n][k].
//   rows_per / node_off select the source (1024, 0) or sink (32, 1024) slice.
//   M is always a multiple of 128 here, so tiles are exact.
// ===========================================================================
__global__ void part_gemm(const float* __restrict__ nodes,
                          const _Float16* __restrict__ WTg,  // 128x128 f16 [n][k]
                          const float* __restrict__ bias,    // may be nullptr
                          float* __restrict__ out,           // M x 128
                          int rows_per, int node_off) {
  extern __shared__ char smem[];
  _Float16* Af16 = (_Float16*)smem;                          // [128][HS]
  _Float16* WTl  = (_Float16*)(smem + 128 * HS * 2);         // [128][HS]

  const int tid  = threadIdx.x;
  const int row0 = blockIdx.x * 128;

  // Stage A rows (f32 -> f16), 4 elements per iteration, coalesced.
  for (int g4 = tid; g4 < 128 * 32; g4 += 256) {
    int r = g4 >> 5, c4 = (g4 & 31) * 4;
    int g = row0 + r;
    int b = g / rows_per, t = g % rows_per;
    const float4 v = *(const float4*)(nodes +
        ((size_t)(b * NODE_T + node_off + t)) * FC + c4);
    v4h h; h[0] = (_Float16)v.x; h[1] = (_Float16)v.y;
           h[2] = (_Float16)v.z; h[3] = (_Float16)v.w;
    *(v4h*)(Af16 + r * HS + c4) = h;
  }
  // Stage pre-transposed f16 weights, 16B vector copies, coalesced.
  for (int g8 = tid; g8 < 128 * 16; g8 += 256) {
    int n = g8 >> 4, k8 = (g8 & 15) * 8;
    *(v8h*)(WTl + n * HS + k8) = *(const v8h*)(WTg + n * 128 + k8);
  }
  __syncthreads();

  const int w    = tid >> 5;
  const int lane = tid & 31;
  const int half = lane >> 4;

  v8f acc[8];
#pragma unroll
  for (int j = 0; j < 8; ++j) acc[j] = (v8f)(0.0f);

#pragma unroll
  for (int ks = 0; ks < 4; ++ks) {
    v16h a = load_a_tile(Af16, w * 16, ks * 32);
    v16h bm[8];
#pragma unroll
    for (int j = 0; j < 8; ++j) bm[j] = load_b_tileT(WTl, j * 16, ks * 32);
#pragma unroll
    for (int j = 0; j < 8; ++j)
      acc[j] = __builtin_amdgcn_wmma_f32_16x16x32_f16(
          false, a, false, bm[j], (short)0, acc[j], false, false);
  }

  // D layout: element v of acc[j] is (M = v + 8*half, N = j*16 + (lane&15))
#pragma unroll
  for (int j = 0; j < 8; ++j) {
    int n = j * 16 + (lane & 15);
    float bn = bias ? bias[n] : 0.0f;
#pragma unroll
    for (int v = 0; v < 8; ++v) {
      int gRow = row0 + w * 16 + v + 8 * half;
      out[(size_t)gRow * FC + n] = acc[j][v] + bn;
    }
  }
}

// ===========================================================================
// Kernel B: per-edge MLP. Block = 4 t-values x 32 s-values = 128 edges.
//   h1   = LN1(relu(srcpart[t] + snkpart[s]))       (f16 into LDS)
//   h2   = LN2(relu(h1 @ W2 + b2))                  (WMMA f16, f32 acc)
//   out  = h2 . W3 + b3
// ===========================================================================
__global__ void edge_kernel(const float* __restrict__ srcpart,   // (B*T0) x 128
                            const float* __restrict__ snkpart,   // (B*TAU) x 128
                            const _Float16* __restrict__ W2Tg,   // 128x128 f16 [n][k]
                            const float* __restrict__ b2,
                            const float* __restrict__ g1, const float* __restrict__ be1,
                            const float* __restrict__ g2, const float* __restrict__ be2,
                            const float* __restrict__ W3,        // 128
                            const float* __restrict__ b3,        // 1
                            float* __restrict__ out) {
  extern __shared__ char smem[];
  float*    sp   = (float*)smem;                     //  4 x 128 f32   (2 KB)
  float*    sk   = (float*)(smem + 2048);            // 32 x 128 f32  (16 KB)
  float*    prm  = (float*)(smem + 18432);           //  6 x 128 f32   (3 KB)
  _Float16* W2T  = (_Float16*)(smem + 21504);        // [128][HS] f16 (34 KB)
  _Float16* hbuf = (_Float16*)(smem + 21504 + 128 * HS * 2);  // [128][HS], reused

  float* g1l  = prm;        float* be1l = prm + 128;
  float* b2l  = prm + 256;  float* g2l  = prm + 384;
  float* be2l = prm + 512;  float* W3l  = prm + 640;

  const int tid = threadIdx.x;
  const int bi  = blockIdx.x / (T0C / 4);   // batch
  const int t0  = (blockIdx.x % (T0C / 4)) * 4;

  // ---- stage LDS ----
  for (int i = tid; i < 4 * 128; i += 256)
    sp[i] = srcpart[((size_t)(bi * T0C + t0) << 7) + i];
  for (int i = tid; i < 32 * 128; i += 256)
    sk[i] = snkpart[((size_t)(bi * TAUC) << 7) + i];
  for (int g8 = tid; g8 < 128 * 16; g8 += 256) {
    int n = g8 >> 4, k8 = (g8 & 15) * 8;
    *(v8h*)(W2T + n * HS + k8) = *(const v8h*)(W2Tg + n * 128 + k8);
  }
  if (tid < 128) {
    g1l[tid]  = g1[tid];  be1l[tid] = be1[tid];
    b2l[tid]  = b2[tid];  g2l[tid]  = g2[tid];
    be2l[tid] = be2[tid]; W3l[tid]  = W3[tid];
  }
  __syncthreads();

  // ---- layer 1: relu(sp+sk) then LayerNorm, f16 into hbuf (2 threads/edge) ----
  {
    const int e = tid >> 1, hh = tid & 1;
    const float* spr = sp + (e >> 5) * 128;
    const float* skr = sk + (e & 31) * 128;
    float sum = 0.f, ss = 0.f;
    for (int f = hh * 64; f < hh * 64 + 64; ++f) {
      float x = fmaxf(spr[f] + skr[f], 0.0f);
      sum += x; ss += x * x;
    }
    sum += __shfl_xor(sum, 1);
    ss  += __shfl_xor(ss, 1);
    float mean = sum * (1.0f / 128.0f);
    float var  = ss * (1.0f / 128.0f) - mean * mean;
    float rs   = rsqrtf(var + 1e-5f);
    _Float16* hrow = hbuf + e * HS;
    for (int f = hh * 64; f < hh * 64 + 64; ++f) {
      float x = fmaxf(spr[f] + skr[f], 0.0f);
      hrow[f] = (_Float16)((x - mean) * rs * g1l[f] + be1l[f]);
    }
  }
  __syncthreads();

  // ---- layer 2: WMMA 16x16x32 f16, 8 n-tiles x 4 k-steps per wave ----
  const int w    = tid >> 5;
  const int lane = tid & 31;
  const int half = lane >> 4;
  const int ln   = lane & 15;

  v8f acc[8];
#pragma unroll
  for (int j = 0; j < 8; ++j) acc[j] = (v8f)(0.0f);

#pragma unroll
  for (int ks = 0; ks < 4; ++ks) {
    v16h a = load_a_tile(hbuf, w * 16, ks * 32);
    v16h bm[8];
#pragma unroll
    for (int j = 0; j < 8; ++j) bm[j] = load_b_tileT(W2T, j * 16, ks * 32);
#pragma unroll
    for (int j = 0; j < 8; ++j)
      acc[j] = __builtin_amdgcn_wmma_f32_16x16x32_f16(
          false, a, false, bm[j], (short)0, acc[j], false, false);
  }

  // bias + relu in place
#pragma unroll
  for (int j = 0; j < 8; ++j) {
    float bn = b2l[j * 16 + ln];
#pragma unroll
    for (int v = 0; v < 8; ++v)
      acc[j][v] = fmaxf(acc[j][v] + bn, 0.0f);
  }

  // LayerNorm2 per row (row m = v + 8*half), reduce over 8 n-tiles + 16 lanes.
  // Wave w only rewrites rows 16w..16w+15 of hbuf, which only wave w reads.
#pragma unroll
  for (int v = 0; v < 8; ++v) {
    float s1 = 0.f, s2 = 0.f;
#pragma unroll
    for (int j = 0; j < 8; ++j) { float x = acc[j][v]; s1 += x; s2 += x * x; }
#pragma unroll
    for (int off = 1; off < 16; off <<= 1) {
      s1 += __shfl_xor(s1, off);
      s2 += __shfl_xor(s2, off);
    }
    float mean = s1 * (1.0f / 128.0f);
    float rs   = rsqrtf(s2 * (1.0f / 128.0f) - mean * mean + 1e-5f);
    int m = w * 16 + v + 8 * half;
#pragma unroll
    for (int j = 0; j < 8; ++j) {
      int n = j * 16 + ln;
      hbuf[m * HS + n] = (_Float16)((acc[j][v] - mean) * rs * g2l[n] + be2l[n]);
    }
  }
  __syncthreads();

  // ---- layer 3: out = h2n . W3 + b3 (2 threads/edge) ----
  {
    const int e = tid >> 1, hh = tid & 1;
    const _Float16* hrow = hbuf + e * HS;
    float dot = 0.f;
    for (int f = hh * 64; f < hh * 64 + 64; ++f)
      dot += (float)hrow[f] * W3l[f];
    dot += __shfl_xor(dot, 1);
    if (hh == 0)
      out[((size_t)(bi * T0C + t0)) * TAUC + e] = dot + b3[0];
  }
}

// ===========================================================================
extern "C" void kernel_launch(void* const* d_in, const int* in_sizes, int n_in,
                              void* d_out, int out_size, void* d_ws, size_t ws_size,
                              hipStream_t stream) {
  const float* nodes = (const float*)d_in[0];
  const float* W1    = (const float*)d_in[1];
  const float* b1    = (const float*)d_in[2];
  const float* g1    = (const float*)d_in[3];
  const float* be1   = (const float*)d_in[4];
  const float* W2    = (const float*)d_in[5];
  const float* b2    = (const float*)d_in[6];
  const float* g2    = (const float*)d_in[7];
  const float* be2   = (const float*)d_in[8];
  const float* W3    = (const float*)d_in[9];
  const float* b3    = (const float*)d_in[10];
  float* out = (float*)d_out;

  // workspace: srcpart | snkpart | f16 weights (W1aT, W1bT, W2T)
  float*    srcpart = (float*)d_ws;                         // 8192 x 128 f32 = 4 MB
  float*    snkpart = srcpart + (size_t)BATCH * T0C * FC;   //  256 x 128 f32 = 128 KB
  _Float16* wT      = (_Float16*)(snkpart + (size_t)BATCH * TAUC * FC); // 3 x 32 KB

  const size_t ldsA = 2u * 128 * HS * 2;                    // Af16 + WT     (68 KB)
  const size_t ldsB = 21504 + 2u * 128 * HS * 2;            // + sp/sk/prm   (89 KB)

  // one-shot weight transpose/convert
  prep_weights<<<3, 256, 0, stream>>>(W1, W2, wT);

  // source @ W1b  (W1 rows 128..255, no bias)
  part_gemm<<<(BATCH * T0C) / 128, 256, ldsA, stream>>>(
      nodes, wT + 1 * 128 * 128, nullptr, srcpart, T0C, 0);
  // sink @ W1a + b1  (W1 rows 0..127)
  part_gemm<<<(BATCH * TAUC) / 128, 256, ldsA, stream>>>(
      nodes, wT + 0 * 128 * 128, b1, snkpart, TAUC, T0C);

  // per-edge MLP: one block per (b, 4 t-values); 2048 blocks x 256 threads
  edge_kernel<<<BATCH * (T0C / 4), 256, ldsB, stream>>>(
      srcpart, snkpart, wT + 2 * 128 * 128, b2, g1, be1, g2, be2, W3, b3, out);
}
